// CombinedGraphTransformer_2241972928607
// MI455X (gfx1250) — matrix operand order repaired and checked by reading, not compiled, not run on hardware
//
#include <hip/hip_runtime.h>
#include <hip/hip_bf16.h>
#include <math.h>

// ---------------- problem constants ----------------
#define Dm   128
#define Hh   4
#define Cc   128
#define Ll   4
#define NCn  65536
#define NPn  32768
#define NEn  65536
#define Bg   1024
#define HC   512     // H*C
#define BW   7
#define SLOPE 0.2f

typedef __attribute__((ext_vector_type(16))) __bf16 v16bf;
typedef __attribute__((ext_vector_type(8)))  float  v8f;
typedef __attribute__((ext_vector_type(4)))  unsigned int v4u;

union Frag { v4u u[2]; v16bf v; };

// ---------------- helpers ----------------
__device__ __forceinline__ unsigned short f2bf(float f) {
    unsigned u = __float_as_uint(f);
    unsigned r = u + 0x7FFFu + ((u >> 16) & 1u);   // round-to-nearest-even
    return (unsigned short)(r >> 16);
}
__device__ __forceinline__ unsigned fmap(float f) {
    unsigned u = __float_as_uint(f);
    return (u & 0x80000000u) ? ~u : (u | 0x80000000u);   // order-preserving map
}
__device__ __forceinline__ float funmap(unsigned u) {
    return __uint_as_float((u & 0x80000000u) ? (u & 0x7FFFFFFFu) : ~u);
}
__device__ __forceinline__ float lrelu(float v) { return v > 0.f ? v : v * SLOPE; }

// ---------------- weight pre-pack: f32 [L,3,K=128,N=512] -> bf16 [L,3,N=512,K=128] ----------------
__global__ void pack_weights_kernel(const float* __restrict__ Wl, const float* __restrict__ Wr,
                                    unsigned short* __restrict__ Wlb, unsigned short* __restrict__ Wrb) {
    const int T = Ll * 3 * HC * Dm;
    int idx = blockIdx.x * blockDim.x + threadIdx.x;
    if (idx >= 2 * T) return;
    const float* src = (idx < T) ? Wl : Wr;
    unsigned short* dst = (idx < T) ? Wlb : Wrb;
    int t = (idx < T) ? idx : idx - T;
    int slice = t / (HC * Dm);
    int rem = t - slice * HC * Dm;
    int n = rem >> 7;      // 0..511
    int k = rem & 127;     // 0..127
    dst[t] = f2bf(src[(size_t)slice * Dm * HC + (size_t)k * HC + n]);
}

// ---------------- embeddings ----------------
__global__ void embed_cell_kernel(const int* __restrict__ cell_x, const float* __restrict__ emb,
                                  float* __restrict__ cell) {
    int idx = blockIdx.x * blockDim.x + threadIdx.x;
    if (idx >= NCn * Dm) return;
    int n = idx >> 7, c = idx & 127;
    cell[idx] = emb[cell_x[n] * Dm + c];
}
__global__ void embed_piece_kernel(const int* __restrict__ piece_x, const float* __restrict__ emb,
                                   unsigned short* __restrict__ piece_bf) {
    int idx = blockIdx.x * blockDim.x + threadIdx.x;
    if (idx >= NPn * Dm) return;
    int p = idx >> 7, c = idx & 127;
    piece_bf[idx] = f2bf(emb[piece_x[p] * Dm + c]);
}
__global__ void cvt_cell_bf_kernel(const float* __restrict__ cell, unsigned short* __restrict__ cell_bf) {
    int idx = blockIdx.x * blockDim.x + threadIdx.x;
    if (idx >= NCn * Dm) return;
    cell_bf[idx] = f2bf(cell[idx]);
}
__global__ void acc_init_kernel(const float* __restrict__ cb, float* __restrict__ acc) {
    // cb points at conv_bias[l, :, :] = [3, C]
    int idx = blockIdx.x * blockDim.x + threadIdx.x;
    if (idx >= NCn * Dm) return;
    int c = idx & 127;
    acc[idx] = cb[c] + cb[Cc + c] + cb[2 * Cc + c];
}

// ---------------- bf16 WMMA GEMM: Out[M,512] = A[M,128] @ Bt^T + bias ----------------
// A: bf16 [M,128] row-major.  Bt: bf16 [512,128] (N-major, i.e. W transposed).
// Register double-buffered over the 4 K-steps so loads of step ks+1 overlap the
// WMMAs of step ks (avoids s_wait_loadcnt 0 before every v_wmma).
__global__ __launch_bounds__(256)
void gemm_bf16_kernel(const unsigned short* __restrict__ A,
                      const unsigned short* __restrict__ Bt,
                      const float* __restrict__ bias,
                      float* __restrict__ Out, int M) {
    const int lane = threadIdx.x & 31;
    const int w    = threadIdx.x >> 5;      // 0..7
    const int mw   = w & 3;                 // 4 M-subtiles
    const int nw   = w >> 2;                // 2 N-subtiles
    const int m0   = blockIdx.x * 64 + mw * 16;
    const int n0   = blockIdx.y * 128 + nw * 64;
    const int r15  = lane & 15;
    const int hi   = lane >> 4;

    const v4u* pa  = (const v4u*)(A  + (size_t)(m0 + r15) * Dm);
    const v4u* pb0 = (const v4u*)(Bt + (size_t)(n0 +  0 + r15) * Dm);
    const v4u* pb1 = (const v4u*)(Bt + (size_t)(n0 + 16 + r15) * Dm);
    const v4u* pb2 = (const v4u*)(Bt + (size_t)(n0 + 32 + r15) * Dm);
    const v4u* pb3 = (const v4u*)(Bt + (size_t)(n0 + 48 + r15) * Dm);

    Frag a[2], b0[2], b1[2], b2[2], b3[2];

    // prologue: fetch k-step 0
    {
        const int ia = 2 * hi;      // A: lane<16 -> K0-15 ; lane>=16 -> K16-31
        const int ib = 2 * hi;      // B: lane<16 -> K0-15 ; lane>=16 -> K16-31
        a[0].u[0]  = pa[ia];      a[0].u[1]  = pa[ia + 1];
        b0[0].u[0] = pb0[ib];     b0[0].u[1] = pb0[ib + 1];
        b1[0].u[0] = pb1[ib];     b1[0].u[1] = pb1[ib + 1];
        b2[0].u[0] = pb2[ib];     b2[0].u[1] = pb2[ib + 1];
        b3[0].u[0] = pb3[ib];     b3[0].u[1] = pb3[ib + 1];
    }

    v8f c0 = {0.f,0.f,0.f,0.f,0.f,0.f,0.f,0.f};
    v8f c1 = c0, c2 = c0, c3 = c0;

    #pragma unroll
    for (int ks = 0; ks < 4; ks++) {
        const int cur = ks & 1;
        const int nxt = cur ^ 1;
        if (ks < 3) {               // prefetch next K-step while current WMMAs run
            const int ia = (ks + 1) * 4 + 2 * hi;
            const int ib = (ks + 1) * 4 + 2 * hi;
            a[nxt].u[0]  = pa[ia];      a[nxt].u[1]  = pa[ia + 1];
            b0[nxt].u[0] = pb0[ib];     b0[nxt].u[1] = pb0[ib + 1];
            b1[nxt].u[0] = pb1[ib];     b1[nxt].u[1] = pb1[ib + 1];
            b2[nxt].u[0] = pb2[ib];     b2[nxt].u[1] = pb2[ib + 1];
            b3[nxt].u[0] = pb3[ib];     b3[nxt].u[1] = pb3[ib + 1];
        }
        c0 = __builtin_amdgcn_wmma_f32_16x16x32_bf16(false, a[cur].v, false, b0[cur].v, (short)0, c0, false, false);
        c1 = __builtin_amdgcn_wmma_f32_16x16x32_bf16(false, a[cur].v, false, b1[cur].v, (short)0, c1, false, false);
        c2 = __builtin_amdgcn_wmma_f32_16x16x32_bf16(false, a[cur].v, false, b2[cur].v, (short)0, c2, false, false);
        c3 = __builtin_amdgcn_wmma_f32_16x16x32_bf16(false, a[cur].v, false, b3[cur].v, (short)0, c3, false, false);
    }

    const float bb0 = bias[n0 +  0 + r15];
    const float bb1 = bias[n0 + 16 + r15];
    const float bb2 = bias[n0 + 32 + r15];
    const float bb3 = bias[n0 + 48 + r15];

    #pragma unroll
    for (int g = 0; g < 8; g++) {
        int row = m0 + g + hi * 8;          // C/D layout: vgpr g, lanes>=16 -> M+8
        size_t base = (size_t)row * HC;
        Out[base + n0 +  0 + r15] = c0[g] + bb0;
        Out[base + n0 + 16 + r15] = c1[g] + bb1;
        Out[base + n0 + 32 + r15] = c2[g] + bb2;
        Out[base + n0 + 48 + r15] = c3[g] + bb3;
    }
}

// ---------------- edge pass 1: scores + segment max ----------------
__global__ void score_kernel(const float* __restrict__ xl, const float* __restrict__ xr,
                             const int* __restrict__ src, const int* __restrict__ dst,
                             const float* __restrict__ att, float* __restrict__ sbuf,
                             unsigned int* __restrict__ mbuf, int E) {
    int idx = blockIdx.x * blockDim.x + threadIdx.x;
    if (idx >= E * Hh) return;
    int e = idx >> 2, h = idx & 3;
    int s = src[e], d = dst[e];
    const float4* pl = (const float4*)(xl + (size_t)s * HC + h * Cc);
    const float4* pr = (const float4*)(xr + (size_t)d * HC + h * Cc);
    const float4* pa = (const float4*)(att + h * Cc);
    float sc = 0.f;
    #pragma unroll 8
    for (int i = 0; i < 32; i++) {
        float4 a = pl[i], b = pr[i], wv = pa[i];
        sc += lrelu(a.x + b.x) * wv.x;
        sc += lrelu(a.y + b.y) * wv.y;
        sc += lrelu(a.z + b.z) * wv.z;
        sc += lrelu(a.w + b.w) * wv.w;
    }
    sbuf[idx] = sc;
    atomicMax(&mbuf[d * Hh + h], fmap(sc));
}

// ---------------- edge pass 2: exp + segment sum ----------------
__global__ void denom_kernel(float* __restrict__ sbuf, const int* __restrict__ dst,
                             const unsigned int* __restrict__ mbuf, float* __restrict__ denom, int E) {
    int idx = blockIdx.x * blockDim.x + threadIdx.x;
    if (idx >= E * Hh) return;
    int e = idx >> 2, h = idx & 3;
    int d = dst[e];
    float p = __expf(sbuf[idx] - funmap(mbuf[d * Hh + h]));
    sbuf[idx] = p;
    atomicAdd(&denom[d * Hh + h], p);
}

// ---------------- edge pass 3: alpha-weighted scatter ----------------
// One thread per EDGE; the 4 heads hit the same output channel, so combine them
// in registers -> 128 atomics/edge instead of 512 (head-mean 1/H folded in).
__global__ void scatter_kernel(const float* __restrict__ sbuf, const float* __restrict__ xl,
                               const int* __restrict__ src, const int* __restrict__ dst,
                               const float* __restrict__ denom, float* __restrict__ acc, int E) {
    int e = blockIdx.x * blockDim.x + threadIdx.x;
    if (e >= E) return;
    int s = src[e], d = dst[e];
    const float a0 = sbuf[e * Hh + 0] / (denom[d * Hh + 0] + 1e-16f) * (1.0f / Hh);
    const float a1 = sbuf[e * Hh + 1] / (denom[d * Hh + 1] + 1e-16f) * (1.0f / Hh);
    const float a2 = sbuf[e * Hh + 2] / (denom[d * Hh + 2] + 1e-16f) * (1.0f / Hh);
    const float a3 = sbuf[e * Hh + 3] / (denom[d * Hh + 3] + 1e-16f) * (1.0f / Hh);
    const float4* p0 = (const float4*)(xl + (size_t)s * HC + 0 * Cc);
    const float4* p1 = (const float4*)(xl + (size_t)s * HC + 1 * Cc);
    const float4* p2 = (const float4*)(xl + (size_t)s * HC + 2 * Cc);
    const float4* p3 = (const float4*)(xl + (size_t)s * HC + 3 * Cc);
    float* po = acc + (size_t)d * Dm;
    #pragma unroll 4
    for (int i = 0; i < 32; i++) {
        float4 x0 = p0[i], x1 = p1[i], x2 = p2[i], x3 = p3[i];
        atomicAdd(po + i * 4 + 0, a0 * x0.x + a1 * x1.x + a2 * x2.x + a3 * x3.x);
        atomicAdd(po + i * 4 + 1, a0 * x0.y + a1 * x1.y + a2 * x2.y + a3 * x3.y);
        atomicAdd(po + i * 4 + 2, a0 * x0.z + a1 * x1.z + a2 * x2.z + a3 * x3.z);
        atomicAdd(po + i * 4 + 3, a0 * x0.w + a1 * x1.w + a2 * x2.w + a3 * x3.w);
    }
}

__global__ void relu_cell_kernel(const float* __restrict__ acc, float* __restrict__ cell) {
    int idx = blockIdx.x * blockDim.x + threadIdx.x;
    if (idx >= NCn * Dm) return;
    cell[idx] = fmaxf(acc[idx], 0.f);
}

// ---------------- pooling + head ----------------
__global__ void pool_kernel(const float* __restrict__ cell, float* __restrict__ pooled) {
    int idx = blockIdx.x * blockDim.x + threadIdx.x;
    if (idx >= Bg * Dm) return;
    int b = idx >> 7, c = idx & 127;
    const int per = NCn / Bg; // 64
    float s = 0.f;
    #pragma unroll 8
    for (int i = 0; i < per; i++) s += cell[((size_t)(b * per + i)) * Dm + c];
    pooled[idx] = s * (1.0f / per);
}

__global__ __launch_bounds__(64)
void head_kernel(const float* __restrict__ pooled,
                 const float* __restrict__ fc1_W, const float* __restrict__ fc1_b,
                 const float* __restrict__ pol_W, const float* __restrict__ pol_b,
                 const float* __restrict__ val_W, const float* __restrict__ val_b,
                 float* __restrict__ out_pol, float* __restrict__ out_val) {
    __shared__ float hsh[64];
    int b = blockIdx.x, j = threadIdx.x;
    float a = fc1_b[j];
    const float* pb = pooled + (size_t)b * Dm;
    #pragma unroll 8
    for (int k = 0; k < Dm; k++) a += pb[k] * fc1_W[k * 64 + j];
    hsh[j] = fmaxf(a, 0.f);
    __syncthreads();
    if (j < BW) {
        float s = pol_b[j];
        #pragma unroll 8
        for (int k = 0; k < 64; k++) s += hsh[k] * pol_W[k * BW + j];
        out_pol[b * BW + j] = s;
    } else if (j == BW) {
        float s = val_b[0];
        #pragma unroll 8
        for (int k = 0; k < 64; k++) s += hsh[k] * val_W[k];
        out_val[b] = tanhf(s);
    }
}

// ---------------- host orchestration ----------------
static inline int cdiv(long long a, int b) { return (int)((a + b - 1) / b); }

extern "C" void kernel_launch(void* const* d_in, const int* in_sizes, int n_in,
                              void* d_out, int out_size, void* d_ws, size_t ws_size,
                              hipStream_t stream) {
    const int*   cell_x    = (const int*)d_in[0];
    const int*   piece_x   = (const int*)d_in[1];
    const int*   occ_src   = (const int*)d_in[2];
    const int*   occ_dst   = (const int*)d_in[3];
    const int*   en_src    = (const int*)d_in[4];
    const int*   en_dst    = (const int*)d_in[5];
    const int*   ee_src    = (const int*)d_in[6];
    const int*   ee_dst    = (const int*)d_in[7];
    // d_in[8] = cell_batch (contiguity assumed: arange // 64)
    const float* cell_emb  = (const float*)d_in[9];
    const float* piece_emb = (const float*)d_in[10];
    const float* Wl        = (const float*)d_in[11];
    const float* bl        = (const float*)d_in[12];
    const float* Wr        = (const float*)d_in[13];
    const float* br        = (const float*)d_in[14];
    const float* att       = (const float*)d_in[15];
    const float* conv_bias = (const float*)d_in[16];
    const float* fc1_W     = (const float*)d_in[17];
    const float* fc1_b     = (const float*)d_in[18];
    const float* pol_W     = (const float*)d_in[19];
    const float* pol_b     = (const float*)d_in[20];
    const float* val_W     = (const float*)d_in[21];
    const float* val_b     = (const float*)d_in[22];

    float* out_pol = (float*)d_out;              // [B,7]
    float* out_val = out_pol + (size_t)Bg * BW;  // [B,1]

    // workspace carving
    char* ws = (char*)d_ws;
    size_t off = 0;
    auto carve = [&](size_t bytes) -> char* {
        char* p = ws + off;
        off += (bytes + 255) & ~(size_t)255;
        return p;
    };
    float*          cell     = (float*)         carve((size_t)NCn * Dm * 4);
    float*          acc      = (float*)         carve((size_t)NCn * Dm * 4);
    float*          xl_buf   = (float*)         carve((size_t)NCn * HC * 4);
    float*          xr_buf   = (float*)         carve((size_t)NCn * HC * 4);
    unsigned short* cell_bf  = (unsigned short*)carve((size_t)NCn * Dm * 2);
    unsigned short* piece_bf = (unsigned short*)carve((size_t)NPn * Dm * 2);
    unsigned short* Wlb      = (unsigned short*)carve((size_t)Ll * 3 * HC * Dm * 2);
    unsigned short* Wrb      = (unsigned short*)carve((size_t)Ll * 3 * HC * Dm * 2);
    float*          sbuf     = (float*)         carve((size_t)NEn * Hh * 4);
    unsigned int*   mbuf     = (unsigned int*)  carve((size_t)NCn * Hh * 4);
    float*          denom    = (float*)         carve((size_t)NCn * Hh * 4);
    float*          pooled   = (float*)         carve((size_t)Bg * Dm * 4);
    (void)ws_size; (void)in_sizes; (void)n_in; (void)out_size;

    const int TB = 256;

    // 1) pack weights to transposed bf16
    {
        int total = 2 * Ll * 3 * HC * Dm;
        pack_weights_kernel<<<cdiv(total, TB), TB, 0, stream>>>(Wl, Wr, Wlb, Wrb);
    }
    // 2) embeddings
    embed_cell_kernel <<<cdiv((long long)NCn * Dm, TB), TB, 0, stream>>>(cell_x, cell_emb, cell);
    embed_piece_kernel<<<cdiv((long long)NPn * Dm, TB), TB, 0, stream>>>(piece_x, piece_emb, piece_bf);

    const int*  srcs[3] = { occ_src, en_src, ee_src };
    const int*  dsts[3] = { occ_dst, en_dst, ee_dst };
    const int   Es[3]   = { NPn, NEn, NEn };

    for (int l = 0; l < Ll; l++) {
        cvt_cell_bf_kernel<<<cdiv((long long)NCn * Dm, TB), TB, 0, stream>>>(cell, cell_bf);
        acc_init_kernel  <<<cdiv((long long)NCn * Dm, TB), TB, 0, stream>>>(conv_bias + (size_t)l * 3 * Cc, acc);

        for (int r = 0; r < 3; r++) {
            const int lr = l * 3 + r;
            const unsigned short* A_src = (r == 0) ? piece_bf : cell_bf;
            const int Ms = (r == 0) ? NPn : NCn;

            // xl = x_src @ Wl + bl ;  xr = cell @ Wr + br
            gemm_bf16_kernel<<<dim3(Ms / 64, HC / 128), TB, 0, stream>>>(
                A_src, Wlb + (size_t)lr * HC * Dm, bl + (size_t)lr * HC, xl_buf, Ms);
            gemm_bf16_kernel<<<dim3(NCn / 64, HC / 128), TB, 0, stream>>>(
                cell_bf, Wrb + (size_t)lr * HC * Dm, br + (size_t)lr * HC, xr_buf, NCn);

            hipMemsetAsync(mbuf, 0, (size_t)NCn * Hh * 4, stream);   // mapped-uint 0 == -inf
            hipMemsetAsync(denom, 0, (size_t)NCn * Hh * 4, stream);

            const int E = Es[r];
            const float* att_lr = att + (size_t)lr * Hh * Cc;
            score_kernel  <<<cdiv((long long)E * Hh, TB), TB, 0, stream>>>(
                xl_buf, xr_buf, srcs[r], dsts[r], att_lr, sbuf, mbuf, E);
            denom_kernel  <<<cdiv((long long)E * Hh, TB), TB, 0, stream>>>(
                sbuf, dsts[r], mbuf, denom, E);
            scatter_kernel<<<cdiv(E, TB), TB, 0, stream>>>(
                sbuf, xl_buf, srcs[r], dsts[r], denom, acc, E);
        }
        relu_cell_kernel<<<cdiv((long long)NCn * Dm, TB), TB, 0, stream>>>(acc, cell);
    }

    pool_kernel<<<cdiv((long long)Bg * Dm, TB), TB, 0, stream>>>(cell, pooled);
    head_kernel<<<Bg, 64, 0, stream>>>(pooled, fc1_W, fc1_b, pol_W, pol_b, val_W, val_b,
                                       out_pol, out_val);
}
